// M2A_49383533969914
// MI455X (gfx1250) — matrix operand order repaired and checked by reading, not compiled, not run on hardware
//
#include <hip/hip_runtime.h>
#include <hip/hip_bf16.h>
#include <stdint.h>

typedef __bf16 bf16_t;
typedef __attribute__((ext_vector_type(16))) __bf16 v16bf;
typedef __attribute__((ext_vector_type(8)))  __bf16 v8bf;
typedef __attribute__((ext_vector_type(4)))  __bf16 v4bf;
typedef __attribute__((ext_vector_type(8)))  float  v8f;

union Frag { v16bf v; v8bf h[2]; };

__device__ __forceinline__ bf16_t f2bf(float f) { return (bf16_t)f; }
__device__ __forceinline__ float  bf2f(bf16_t b) { return (float)b; }

__device__ __forceinline__ v8f wmma_bf16(v16bf a, v16bf b, v8f c) {
    return __builtin_amdgcn_wmma_f32_16x16x32_bf16(
        false, a, false, b, (short)0, c, false, false);
}

// A-fragment (16x32 bf16): lanes 0-15 take K 0-15, lanes 16-31 K 16-31.
__device__ __forceinline__ v16bf lds_afrag(const bf16_t* rowbase, int lane) {
    int koff = (lane < 16) ? 0 : 16;
    Frag f;
    f.h[0] = *(const v8bf*)(rowbase + koff);
    f.h[1] = *(const v8bf*)(rowbase + koff + 8);
    return f.v;
}

// B-fragment: 16 consecutive bf16 K values (32B) at p (works for LDS or global).
__device__ __forceinline__ Frag load_bfrag_bf16(const bf16_t* p) {
    Frag f;
    f.h[0] = *(const v8bf*)(p);
    f.h[1] = *(const v8bf*)(p + 8);
    return f;
}

__device__ __forceinline__ Frag load_bfrag_f32(const float* p) {
    Frag f;
#pragma unroll
    for (int q = 0; q < 4; ++q) {
        float4 t = ((const float4*)p)[q];
        f.v[4*q+0] = f2bf(t.x); f.v[4*q+1] = f2bf(t.y);
        f.v[4*q+2] = f2bf(t.z); f.v[4*q+3] = f2bf(t.w);
    }
    return f;
}

// gfx1250 async DMA: copy 16B global -> LDS, tracked by ASYNCcnt.
__device__ __forceinline__ void async_lds_b128(uint32_t lds_byte, uint32_t goff_byte,
                                               const void* sbase) {
    asm volatile("global_load_async_to_lds_b128 %0, %1, %2"
                 :: "v"(lds_byte), "v"(goff_byte), "s"(sbase) : "memory");
}
__device__ __forceinline__ void wait_async0() {
    asm volatile("s_wait_asynccnt 0" ::: "memory");
}

// One-time weight conversion into workspace: Wd (128x512) and Wu (512x128).
__global__ __launch_bounds__(256)
void prep_weights_kernel(const float* __restrict__ Wd, const float* __restrict__ Wu,
                         bf16_t* __restrict__ WdB, bf16_t* __restrict__ WuB)
{
    int i = blockIdx.x * 256 + threadIdx.x;
    float4 a = ((const float4*)Wd)[i];
    float4 b = ((const float4*)Wu)[i];
    ((v4bf*)WdB)[i] = v4bf{f2bf(a.x), f2bf(a.y), f2bf(a.z), f2bf(a.w)};
    ((v4bf*)WuB)[i] = v4bf{f2bf(b.x), f2bf(b.y), f2bf(b.z), f2bf(b.w)};
}

template <bool PRE>
__global__ __launch_bounds__(256)
void tsm_fused_kernel(const float* __restrict__ X,
                      const float* __restrict__ Wd, const float* __restrict__ bd,
                      const float* __restrict__ Wu, const float* __restrict__ bu,
                      const bf16_t* __restrict__ WdB, const bf16_t* __restrict__ WuB,
                      float* __restrict__ OUT)
{
    constexpr int C = 512, H = 56, W = 56, Tt = 8, HW = H * W;
    const int tid  = threadIdx.x;
    const int lane = tid & 31;
    const int wv   = tid >> 5;
    const int nl16 = lane & 15;
    const int mhalf = (lane >= 16) ? 8 : 0;

    const int g  = blockIdx.x;
    const int wc = g % 7;              // 8-wide w chunk
    const int hh = (g / 7) % H;
    const int bb = g / (7 * H);
    const int w0 = wc * 8;

    __shared__ bf16_t sPool[2 * 128 * 32];  // A staging (double buf, 16 KB); scores later
    __shared__ bf16_t sW[2 * 128 * 32];     // async weight K-chunk staging (16 KB)
    __shared__ bf16_t sQ[64 * 128];         // xn(h+1) -> Q -> gate staging (16 KB)
    __shared__ bf16_t sHF[64 * 128];        // h (+ attn residual) for row hh (16 KB)
    bf16_t* sS = sPool;                     // phase-3 scores alias (4 KB)

    // ================= Phase 1: conv_down GEMM  M=128 (2 rows x 8t x 8w), N=128, K=512
    v8f acc[8];
#pragma unroll
    for (int nt = 0; nt < 8; ++nt) acc[nt] = v8f{0.f,0.f,0.f,0.f,0.f,0.f,0.f,0.f};

    const int mt = wv;
    const uint32_t arow_off = (uint32_t)((mt * 16 + nl16) * 32);

    // x staging descriptors: clamped 32-bit offsets + 0/1 row mask.
    uint32_t xoff[4]; uint32_t loff[4]; float msk[4];
#pragma unroll
    for (int i = 0; i < 4; ++i) {
        int idx = i * 256 + tid;
        int k = idx >> 5, gm = idx & 31;
        int m = gm * 4;
        int w = m & 7, t = (m >> 3) & 7, rr = m >> 6;   // rr=1 -> row hh+1
        int hr = (hh + rr < H) ? (hh + rr) : (H - 1);
        msk[i]  = (hh + rr < H) ? 1.f : 0.f;
        xoff[i] = (uint32_t)(((bb * Tt + t) * C + k) * HW + hr * W + (w0 + w));
        loff[i] = (uint32_t)(m * 32 + k);
    }

    auto stage = [&](int kcn, bf16_t* buf) {            // fp32 -> bf16 LDS staging of x
#pragma unroll
        for (int i = 0; i < 4; ++i) {
            float4 v = *(const float4*)(X + xoff[i]);
            bf16_t* sp = buf + loff[i];
            sp[0]  = f2bf(v.x * msk[i]);
            sp[32] = f2bf(v.y * msk[i]);
            sp[64] = f2bf(v.z * msk[i]);
            sp[96] = f2bf(v.w * msk[i]);
            xoff[i] += 32u * HW;
            if (kcn < 15) __builtin_prefetch(X + xoff[i], 0, 3);
        }
    };

    // Async DMA of one 8 KB weight K-chunk into LDS (whole workgroup: 2x16B each).
    const int wn = tid >> 1, whf = tid & 1;
    const uint32_t wgo0 = (uint32_t)(wn * 1024 + whf * 32);       // bytes in WdB row
    const uint32_t wlo0 = (uint32_t)(uintptr_t)sW + (uint32_t)(wn * 64 + whf * 32);
    auto stage_w = [&](int kcn) {
        uint32_t go = wgo0 + (uint32_t)kcn * 64u;
        uint32_t lo = wlo0 + (uint32_t)((kcn & 1) << 13);         // 8 KB buffer select
        async_lds_b128(lo,      go,      WdB);
        async_lds_b128(lo + 16, go + 16, WdB);
    };

    if constexpr (PRE) stage_w(0);
    stage(0, sPool);

    const float* wdf = PRE ? nullptr : (Wd + nl16 * 512 + ((lane < 16) ? 0 : 16));

    for (int kc = 0; kc < 16; ++kc) {
        if constexpr (PRE) wait_async0();   // my chunk-kc weight DMAs done
        __syncthreads();                    // all staging for chunk kc visible
        bf16_t* cur = sPool + ((kc & 1) << 12);
        bf16_t* nxt = sPool + (((kc + 1) & 1) << 12);
        if (kc < 15) {
            if constexpr (PRE) stage_w(kc + 1);
            stage(kc + 1, nxt);
        }

        v16bf a = lds_afrag(cur + arow_off, lane);
        Frag bfr[8];
        if constexpr (PRE) {                // B frags from LDS (low latency, broadcast)
            const bf16_t* swc = sW + ((kc & 1) << 12) + nl16 * 32 + ((lane < 16) ? 0 : 16);
#pragma unroll
            for (int nt = 0; nt < 8; ++nt) bfr[nt] = load_bfrag_bf16(swc + nt * 512);
        } else {
            const float* wb = wdf + kc * 32;
#pragma unroll
            for (int nt = 0; nt < 8; ++nt) bfr[nt] = load_bfrag_f32(wb + nt * 8192);
        }
#pragma unroll
        for (int nt = 0; nt < 8; ++nt) acc[nt] = wmma_bf16(a, bfr[nt].v, acc[nt]);
    }
    // bias
#pragma unroll
    for (int nt = 0; nt < 8; ++nt) {
        float bv = bd[nt * 16 + nl16];
#pragma unroll
        for (int v = 0; v < 8; ++v) acc[nt][v] += bv;
    }

    // ================= Phase 2: LayerNorm; waves 4-7 publish xn(h+1), waves 0-3 keep regs
    float mus[8], rss[8];
#pragma unroll
    for (int v = 0; v < 8; ++v) {
        float s1 = 0.f, s2 = 0.f;
#pragma unroll
        for (int nt = 0; nt < 8; ++nt) { float x = acc[nt][v]; s1 += x; s2 += x * x; }
#pragma unroll
        for (int m = 1; m <= 8; m <<= 1) {
            s1 += __shfl_xor(s1, m, 32);
            s2 += __shfl_xor(s2, m, 32);
        }
        float mu  = s1 * (1.f / 128.f);
        float var = s2 * (1.f / 128.f) - mu * mu;
        float rs  = rsqrtf(var + 1e-5f);
        mus[v] = mu; rss[v] = rs;
        int row = mt * 16 + v + mhalf;
#pragma unroll
        for (int nt = 0; nt < 8; ++nt) {
            int n = nt * 16 + nl16;
            if (mt >= 4) sQ[(row - 64) * 128 + n] = f2bf((acc[nt][v] - mu) * rs);
            else         sHF[row * 128 + n]       = f2bf(acc[nt][v]);
        }
    }
    __syncthreads();

    // ================= Phase 2.5: waves 0-3: Q[m] = xn(h+1)[m] - xn(h)[m] in place
    if (mt < 4) {
#pragma unroll
        for (int v = 0; v < 8; ++v) {
            int row = mt * 16 + v + mhalf;
#pragma unroll
            for (int nt = 0; nt < 8; ++nt) {
                int ii = row * 128 + nt * 16 + nl16;
                float xn = (acc[nt][v] - mus[v]) * rss[v];
                sQ[ii] = f2bf(bf2f(sQ[ii]) - xn);
            }
        }
    }
    __syncthreads();

    // ================= Phase 3: per-pixel T=8 attention (4 pixel pairs as 16x16 tiles)
    const int p    = wv >> 1;
    const int half = wv & 1;
    auto qrow = [&](int i) { return ((i & 7) << 3) + 2 * p + (i >> 3); };

    if (half == 0) {   // scores + masked softmax, one wave per pair
        v8f sc = v8f{0.f,0.f,0.f,0.f,0.f,0.f,0.f,0.f};
#pragma unroll
        for (int kcc = 0; kcc < 4; ++kcc) {
            const bf16_t* rp = sQ + qrow(nl16) * 128 + kcc * 32;
            v16bf a = lds_afrag(rp, lane);
            Frag fb;                                   // B(k=c, n=s) = q[s][c]
            const bf16_t* bp = rp + ((lane < 16) ? 0 : 16);
            fb.h[0] = *(const v8bf*)(bp);
            fb.h[1] = *(const v8bf*)(bp + 8);
            sc = wmma_bf16(a, fb.v, sc);
        }
        const float scale = 0.0883883476483184f;       // 1/sqrt(128)
        bool valid = (((lane >> 3) & 1) == ((lane >> 4) & 1));  // block-diagonal mask
#pragma unroll
        for (int v = 0; v < 8; ++v) {
            float val = valid ? sc[v] * scale : -3.0e38f;
            float mx = val;
#pragma unroll
            for (int m = 1; m <= 4; m <<= 1) mx = fmaxf(mx, __shfl_xor(mx, m, 32));
            float e = __expf(val - mx);
            float s = e;
#pragma unroll
            for (int m = 1; m <= 4; m <<= 1) s += __shfl_xor(s, m, 32);
            float r = valid ? (e / s) : 0.f;
            int Mr = v + mhalf;
            sS[p * 512 + Mr * 32 + nl16]      = f2bf(r);
            sS[p * 512 + Mr * 32 + nl16 + 16] = f2bf(0.f);   // zero K pad
        }
    }
    __syncthreads();

    {   // attn = scores @ q, residual-add into h
        const bf16_t* sp = sS + p * 512 + nl16 * 32;
        v16bf a = lds_afrag(sp, lane);
#pragma unroll
        for (int j = 0; j < 4; ++j) {
            int nt = half * 4 + j;
            Frag fb;
            if (lane < 16) {                 // K = s = 0..15 (rows of q)
#pragma unroll
                for (int s = 0; s < 16; ++s)
                    fb.v[s] = sQ[qrow(s) * 128 + nt * 16 + nl16];
            } else {
#pragma unroll
                for (int s = 0; s < 16; ++s) fb.v[s] = f2bf(0.f);
            }
            v8f d = wmma_bf16(a, fb.v, v8f{0.f,0.f,0.f,0.f,0.f,0.f,0.f,0.f});
#pragma unroll
            for (int v = 0; v < 8; ++v) {
                int hr = qrow(v + mhalf);
                int ii = hr * 128 + nt * 16 + nl16;
                sHF[ii] = f2bf(bf2f(sHF[ii]) + d[v]);
            }
        }
    }
    __syncthreads();

    // ================= Phase 4: conv_up (M=64, N=512, K=128) + sigmoid gate + residual out
    const int mt4 = wv & 3;
    const int nh4 = wv >> 2;
    bf16_t* sUP = sQ;                         // Q is dead: reuse as gate staging

    const int w_o = tid & 7, t_o = (tid >> 3) & 7, cl0 = tid >> 6;
    const uint32_t obase0 =
        (uint32_t)(((bb * Tt + t_o) * C + cl0) * HW + hh * W + w0 + w_o);
    const uint32_t lbase0 = (uint32_t)((t_o * 8 + w_o) * 128 + cl0);

    for (int nchunk = 0; nchunk < 4; ++nchunk) {
        v8f up[4];
#pragma unroll
        for (int j = 0; j < 4; ++j) up[j] = v8f{0.f,0.f,0.f,0.f,0.f,0.f,0.f,0.f};
#pragma unroll
        for (int kc = 0; kc < 4; ++kc) {
            const bf16_t* rp = sHF + (mt4 * 16 + nl16) * 128 + kc * 32;
            v16bf a = lds_afrag(rp, lane);
            int kg = kc * 32 + ((lane < 16) ? 0 : 16);
            Frag bfr[4];
            if constexpr (PRE) {
                const bf16_t* wb = WuB + (size_t)(nchunk * 128 + nh4 * 64 + nl16) * 128 + kg;
#pragma unroll
                for (int j = 0; j < 4; ++j) bfr[j] = load_bfrag_bf16(wb + j * 2048);
            } else {
                const float* wb = Wu + (size_t)(nchunk * 128 + nh4 * 64 + nl16) * 128 + kg;
#pragma unroll
                for (int j = 0; j < 4; ++j) bfr[j] = load_bfrag_f32(wb + j * 2048);
            }
#pragma unroll
            for (int j = 0; j < 4; ++j) up[j] = wmma_bf16(a, bfr[j].v, up[j]);
        }
#pragma unroll
        for (int j = 0; j < 4; ++j) {
            int nloc = (nh4 * 4 + j) * 16 + nl16;
            float bv = bu[nchunk * 128 + nloc];
#pragma unroll
            for (int v = 0; v < 8; ++v) {
                int row = mt4 * 16 + v + mhalf;
                float gt = 1.f / (1.f + __expf(-(up[j][v] + bv)));
                sUP[row * 128 + nloc] = f2bf(gt);
            }
        }
        __syncthreads();
        uint32_t obase = obase0 + (uint32_t)nchunk * 128u * HW;
#pragma unroll
        for (int i = 0; i < 32; ++i) {
            float gt = bf2f(sUP[lbase0 + i * 4]);
            uint32_t off = obase + (uint32_t)i * 4u * HW;
            float xv = __builtin_nontemporal_load(X + off);        // last use of x
            __builtin_nontemporal_store(gt * xv + xv, OUT + off);  // write-only stream
        }
        __syncthreads();
    }
}

extern "C" void kernel_launch(void* const* d_in, const int* in_sizes, int n_in,
                              void* d_out, int out_size, void* d_ws, size_t ws_size,
                              hipStream_t stream) {
    (void)in_sizes; (void)n_in; (void)out_size;
    const float* x  = (const float*)d_in[0];
    const float* Wd = (const float*)d_in[1];
    const float* bd = (const float*)d_in[2];
    const float* Wu = (const float*)d_in[3];
    const float* bu = (const float*)d_in[4];
    float* out = (float*)d_out;
    dim3 grid(8 * 56 * 7);   // (b, h, w/8) workgroups

    const size_t kWBytes = (size_t)2 * 65536 * sizeof(bf16_t);  // 256 KB
    if (ws_size >= kWBytes) {
        bf16_t* WdB = (bf16_t*)d_ws;
        bf16_t* WuB = WdB + 65536;
        prep_weights_kernel<<<64, 256, 0, stream>>>(Wd, Wu, WdB, WuB);
        tsm_fused_kernel<true><<<grid, 256, 0, stream>>>(x, Wd, bd, Wu, bu, WdB, WuB, out);
    } else {
        tsm_fused_kernel<false><<<grid, 256, 0, stream>>>(x, Wd, bd, Wu, bu,
                                                          nullptr, nullptr, out);
    }
}
